// Hypernet_80728205296156
// MI455X (gfx1250) — matrix operand order repaired and checked by reading, not compiled, not run on hardware
//
#include <hip/hip_runtime.h>
#include <hip/hip_bf16.h>

// ---------------------------------------------------------------------------
// MI455X (gfx1250) implementation.
//
// Kernel 1 (lstm_hypernet_kernel): single workgroup, 512 thr = 16 waves on one
//   WGP. W_hh is held RESIDENT IN VGPRs as WMMA A-operands: each wave owns a
//   32x128 slab = 8 v16bf chunks (64 VGPRs/lane), loaded once per phase from a
//   pre-swizzled bf16 LDS staging buffer. Per step: g = W_hh @ h computed with
//   8 v_wmma_f32_16x16x32_bf16 (B = h broadcast into all 16 columns; all four
//   32B B chunks prefetched first so the serial chain pays one LDS latency),
//   column 0 extracted to LDS, gate math on threads 0..127, wave-shuffle
//   reduction for hs@fc1. The generated conv weight is emitted pre-swizzled
//   into the conv's WMMA-A layout.
//
// Kernel 2 (conv_wmma_kernel): implicit GEMM, M=32 outch x N=224 cols x K=784.
//   Block = 448 thr (14 waves), one (n,h) row per block. Weights staged to LDS
//   via the Tensor Data Mover (tensor_load_to_lds, TENSORcnt) overlapped with
//   the f32->bf16 input staging; 25 K-steps x 2 v_wmma_f32_16x16x32_bf16/wave.
//   Roofline: 161 GFLOP bf16 (~30us) vs 616 MB HBM (~26us) -> balanced.
// ---------------------------------------------------------------------------

extern __shared__ char dyn_smem[];

typedef __attribute__((ext_vector_type(16))) __bf16        v16bf;
typedef __attribute__((ext_vector_type(8)))  float         v8f;
typedef __attribute__((ext_vector_type(4)))  unsigned int  v4u;
typedef __attribute__((ext_vector_type(4)))  int           v4i;
typedef __attribute__((ext_vector_type(8)))  int           v8i;

__device__ __forceinline__ unsigned short f32_to_bf16(float f) {
    unsigned int u = __float_as_uint(f);
    u += 0x7FFFu + ((u >> 16) & 1u);        // round-to-nearest-even
    return (unsigned short)(u >> 16);
}

__device__ __forceinline__ float sigm(float x) { return 1.f / (1.f + __expf(-x)); }

#define HID 128
#define G4  512
#define WA_HALFS 25600          // 25 ksteps * 2 mtiles * 32 lanes * 16 halfs

#if __has_builtin(__builtin_amdgcn_tensor_load_to_lds) && \
    __has_builtin(__builtin_amdgcn_s_wait_tensorcnt)
#define USE_TDM 1
#endif

// ---------------------------------------------------------------------------
// Kernel 1: LSTM hypernetwork, recurrent matvec on the matrix unit.
// grid=1, block=512, dyn LDS = 139024 B.
//
// LDS byte map:
//   [0,131072)        Wbf   : W_hh bf16, pre-swizzled per-lane WMMA A chunks
//   [131072,131328)   hbf   : h as bf16 (128)   (32B aligned -> B chunk loads)
//   [131328,133376)   g     : 512 f32 gate pre-activations
//   [133376,135424)   win   : 512 f32 W_ih[:,0]
//   [135424,137472)   bia   : 512 f32 b_ih+b_hh
//   [137472,137984)   fc    : 128 f32
//   [137984,138496)   ps    : 128 f32 p
//   [138496,139008)   hseq  : 128 f32 h1_seq
//   [139008,139024)   red   : 4 f32 wave partials
// ---------------------------------------------------------------------------
__global__ __launch_bounds__(512)
void lstm_hypernet_kernel(const float* __restrict__ p,
                          const float* __restrict__ W_ih1, const float* __restrict__ W_hh1,
                          const float* __restrict__ b_ih1, const float* __restrict__ b_hh1,
                          const float* __restrict__ W_ih2, const float* __restrict__ W_hh2,
                          const float* __restrict__ b_ih2, const float* __restrict__ b_hh2,
                          const float* __restrict__ fc1,
                          unsigned short* __restrict__ wA)
{
    char* sm = dyn_smem;
    unsigned short* Wbf  = (unsigned short*)sm;
    unsigned short* hbf  = (unsigned short*)(sm + 131072);
    float* g    = (float*)(sm + 131328);
    float* win  = (float*)(sm + 133376);
    float* bia  = (float*)(sm + 135424);
    float* fc   = (float*)(sm + 137472);
    float* ps   = (float*)(sm + 137984);
    float* hseq = (float*)(sm + 138496);
    float* red  = (float*)(sm + 139008);

    const int t    = threadIdx.x;
    const int w    = t >> 5;          // wave id: owns output rows [w*32, w*32+32)
    const int lane = t & 31;
    const int hi   = lane >> 4;
    const int nn   = lane & 15;

    // zero swizzled conv-weight buffer (tap-49 pad slots must stay 0)
    for (int i = t; i < WA_HALFS / 2; i += 512) ((unsigned int*)wA)[i] = 0u;
    if (t < HID) { fc[t] = fc1[t]; ps[t] = p[t]; }

    for (int phase = 0; phase < 2; ++phase) {
        const float* Wg = phase ? W_hh2 : W_hh1;
        win[t] = phase ? W_ih2[t] : W_ih1[t];
        bia[t] = phase ? (b_ih2[t] + b_hh2[t]) : (b_ih1[t] + b_hh1[t]);

        // Scatter W_hh -> bf16 WMMA A-operand layout (coalesced global reads).
        // chunk (w,mt,kt) lane L=(hi,m) half i=(gbit,tsub):
        //   element = W_hh[w*32 + mt*16 + m][kt*32 + hi*16 + gbit*8 + tsub]
        for (int i = t; i < G4 * HID; i += 512) {
            int row = i >> 7, k = i & 127;
            int ww = row >> 5, mt = (row >> 4) & 1, m = row & 15;
            int kt = k >> 5, kr = k & 31;
            int gb = kr >> 4, rem = kr & 15;
            int hh = rem >> 3, ts = rem & 7;
            int dst = (((ww * 8 + mt * 4 + kt) * 32) + (gb * 16 + m)) * 16 + (hh * 8 + ts);
            Wbf[dst] = f32_to_bf16(Wg[i]);
        }
        if (t < HID) hbf[t] = 0;      // h = 0
        float c = 0.f;
        __syncthreads();

        // pull this wave's W slab into VGPRs (stays resident the whole phase)
        v16bf Wreg[8];
        const v16bf* wvp = (const v16bf*)Wbf;
        #pragma unroll
        for (int j = 0; j < 8; ++j) Wreg[j] = wvp[(w * 8 + j) * 32 + lane];
        __syncthreads();

        const unsigned short* hb = hbf + hi * 16;   // per-half-wave broadcast base
        const int nsteps = phase ? (HID + 25088) : 256;
        for (int s = 0; s < nsteps; ++s) {
            float x;
            if (phase == 0)      x = ps[s & (HID - 1)];
            else if (s < HID)    x = hseq[s];
            else                 x = ps[(s - HID) & (HID - 1)];

            // prefetch ALL B chunks first: one LDS round-trip on the chain
            v16bf bch[4];
            #pragma unroll
            for (int kt = 0; kt < 4; ++kt)
                bch[kt] = *(const v16bf*)(hb + kt * 32);

            // g_slab = W_slab @ h : 8 WMMAs, two independent accumulator chains
            v8f a0 = {0.f, 0.f, 0.f, 0.f, 0.f, 0.f, 0.f, 0.f};
            v8f a1 = a0;
            #pragma unroll
            for (int kt = 0; kt < 4; ++kt) {
                a0 = __builtin_amdgcn_wmma_f32_16x16x32_bf16(false, Wreg[kt],     false, bch[kt],
                                                             (short)0, a0, false, false);
                a1 = __builtin_amdgcn_wmma_f32_16x16x32_bf16(false, Wreg[4 + kt], false, bch[kt],
                                                             (short)0, a1, false, false);
            }
            // column n=0 (lanes 0 and 16) holds the matvec result
            if (nn == 0) {
                float* gp = g + w * 32 + hi * 8;
                #pragma unroll
                for (int r = 0; r < 8; ++r) { gp[r] = a0[r]; gp[16 + r] = a1[r]; }
            }
            __syncthreads();

            if (t < HID) {
                float gi = g[t]       + win[t]       * x + bia[t];
                float gf = g[t + 128] + win[t + 128] * x + bia[t + 128];
                float gg = g[t + 256] + win[t + 256] * x + bia[t + 256];
                float go = g[t + 384] + win[t + 384] * x + bia[t + 384];
                c = sigm(gf) * c + sigm(gi) * tanhf(gg);
                float hn = sigm(go) * tanhf(c);
                hbf[t] = f32_to_bf16(hn);
                float prod = hn * fc[t];                  // hs @ fc1, wave32 reduce
                prod += __shfl_down(prod, 16, 32);
                prod += __shfl_down(prod,  8, 32);
                prod += __shfl_down(prod,  4, 32);
                prod += __shfl_down(prod,  2, 32);
                prod += __shfl_down(prod,  1, 32);
                if ((t & 31) == 0) red[t >> 5] = prod;
            }
            __syncthreads();

            if (t == 0) {
                float wv_ = red[0] + red[1] + red[2] + red[3];
                if (phase == 0) {
                    if (s >= HID) hseq[s - HID] = wv_;    // h1_seq
                } else if (s >= HID) {
                    // scatter generated weight into conv's WMMA-A swizzled layout
                    int sw  = s - HID;
                    int oc  = sw / 784;
                    int rem = sw - oc * 784;
                    int ch  = rem / 49;
                    int tap = rem - ch * 49;
                    int kk  = tap >> 1, gb = tap & 1;
                    int mt  = oc >> 4,  m  = oc & 15;
                    int hh  = ch >> 3,  tl = ch & 7;
                    int idx = ((kk * 2 + mt) * 32 + (gb * 16 + m)) * 16 + ch;
                    wA[idx] = f32_to_bf16(wv_);
                }
            }
            __syncthreads();
        }
        __syncthreads();
    }
}

// ---------------------------------------------------------------------------
// Kernel 2: implicit-GEMM conv via bf16 WMMA + TDM weight staging.
// grid = (224 rows, 64 batches), block = 448 (14 waves), dyn LDS = 110080 B.
// ---------------------------------------------------------------------------
#define CONV_THREADS 448
#define TILE_COLS 230                        // 224 + 6 halo
#define IN_TILE_HALFS (8 * TILE_COLS * 16)   // 8th row zero pad (tap 49)
#define WT_LDS_OFF (IN_TILE_HALFS * 2)       // 58880: weight region byte offset

__global__ __launch_bounds__(CONV_THREADS)
void conv_wmma_kernel(const float* __restrict__ input,
                      const unsigned short* __restrict__ wA,
                      float* __restrict__ out)
{
    char* sm = dyn_smem;
    unsigned short* in_t = (unsigned short*)sm;                  // [8][230][16] bf16
    unsigned short* w_t  = (unsigned short*)(sm + WT_LDS_OFF);   // 25600 bf16

    const int tid = threadIdx.x;
    const int h   = blockIdx.x;
    const int n   = blockIdx.y;

#ifdef USE_TDM
    // Tensor Data Mover: DMA the 51200 B pre-swizzled weight block into LDS.
    // D# (ISA 08 §8): 1-D tile of 12800 dwords. Issued by wave 0 only,
    // overlapped with the input-staging conversion loop below.
    if (tid < 32) {
        unsigned long long ga = (unsigned long long)(uintptr_t)wA;
        v4u g0; v8i g1; v4i g2 = {0, 0, 0, 0}, g3 = {0, 0, 0, 0};
        g0[0] = 1u;                                    // count=1, user mode
        g0[1] = (unsigned)WT_LDS_OFF;                  // lds_addr
        g0[2] = (unsigned)(ga & 0xffffffffu);          // global_addr[31:0]
        g0[3] = (unsigned)((ga >> 32) & 0x01ffffffu)   // global_addr[56:32]
              | (2u << 30);                            // type=2 (image)
        g1[0] = 0x00020000;                            // data_size=4B, mask=0
        g1[1] = (int)(12800u << 16);                   // tensor_dim0[15:0]=12800
        g1[2] = 0;                                     // tensor_dim0[31:16]=0
        g1[3] = (int)(12800u << 16);                   // tile_dim0=12800
        g1[4] = 0;                                     // tile_dim1/2 unused
        g1[5] = 12800;                                 // tensor_dim0_stride
        g1[6] = 0; g1[7] = 0;
#if defined(__clang_major__) && __clang_major__ >= 23
        v8i g4 = {0, 0, 0, 0, 0, 0, 0, 0};
        __builtin_amdgcn_tensor_load_to_lds(g0, g1, g2, g3, g4, 0);
#else
        __builtin_amdgcn_tensor_load_to_lds(g0, g1, g2, g3, 0);
#endif
    }
#else
    {
        const uint4* src = (const uint4*)wA;
        uint4*       dst = (uint4*)w_t;
        for (int i = tid; i < WA_HALFS / 8; i += CONV_THREADS) dst[i] = src[i];
    }
#endif

    // stage input row-strip, f32 -> bf16, layout [row][col][ch]
    for (int idx = tid; idx < IN_TILE_HALFS; idx += CONV_THREADS) {
        int ch  = idx / (8 * TILE_COLS);
        int rem = idx - ch * (8 * TILE_COLS);
        int r   = rem / TILE_COLS;
        int col = rem - r * TILE_COLS;
        float v = 0.f;
        if (r < 7) {
            int iy = h + r - 3;
            int ix = col - 3;
            if (iy >= 0 && iy < 224 && ix >= 0 && ix < 224)
                v = input[(((long)n * 16 + ch) * 224 + iy) * 224 + ix];
        }
        in_t[(r * TILE_COLS + col) * 16 + ch] = f32_to_bf16(v);
    }
#ifdef USE_TDM
    if (tid < 32) __builtin_amdgcn_s_wait_tensorcnt(0);
#endif
    __syncthreads();

    const int lane = tid & 31;
    const int wv   = tid >> 5;
    const int hi   = lane >> 4;
    const int nn   = lane & 15;
    const int wcol = wv * 16 + nn;

    const v16bf* wvec = (const v16bf*)w_t;
    const v16bf* ivec = (const v16bf*)in_t;

    v8f acc0 = {0.f, 0.f, 0.f, 0.f, 0.f, 0.f, 0.f, 0.f};
    v8f acc1 = acc0;

    #pragma unroll
    for (int kk = 0; kk < 25; ++kk) {
        v16bf a0 = wvec[(kk * 2 + 0) * 32 + lane];     // oc 0..15
        v16bf a1 = wvec[(kk * 2 + 1) * 32 + lane];     // oc 16..31
        int tap = 2 * kk + hi;
        int ky  = tap / 7;
        int kx  = tap - ky * 7;                        // tap 49 -> zeroed pad row
        v16bf b = ivec[ky * TILE_COLS + wcol + kx];
        acc0 = __builtin_amdgcn_wmma_f32_16x16x32_bf16(false, a0, false, b,
                                                       (short)0, acc0, false, false);
        acc1 = __builtin_amdgcn_wmma_f32_16x16x32_bf16(false, a1, false, b,
                                                       (short)0, acc1, false, false);
    }

    long base = (long)n * 32 * 50176 + (long)h * 224;
    #pragma unroll
    for (int r = 0; r < 8; ++r) {
        int oc0 = hi * 8 + r;
        int oc1 = 16 + hi * 8 + r;
        out[base + (long)oc0 * 50176 + wcol] = acc0[r];
        out[base + (long)oc1 * 50176 + wcol] = acc1[r];
    }
}

// ---------------------------------------------------------------------------
extern "C" void kernel_launch(void* const* d_in, const int* in_sizes, int n_in,
                              void* d_out, int out_size, void* d_ws, size_t ws_size,
                              hipStream_t stream) {
    const float* p      = (const float*)d_in[0];
    const float* input  = (const float*)d_in[1];
    const float* W_ih1  = (const float*)d_in[2];
    const float* W_hh1  = (const float*)d_in[3];
    const float* b_ih1  = (const float*)d_in[4];
    const float* b_hh1  = (const float*)d_in[5];
    const float* W_ih2  = (const float*)d_in[6];
    const float* W_hh2  = (const float*)d_in[7];
    const float* b_ih2  = (const float*)d_in[8];
    const float* b_hh2  = (const float*)d_in[9];
    const float* fc1    = (const float*)d_in[10];
    float*       out    = (float*)d_out;
    unsigned short* wA  = (unsigned short*)d_ws;   // 51200 B swizzled bf16 weights

    size_t smem1 = 139024;   // Wbf + hbf + g + win + bia + fc + ps + hseq + red
    lstm_hypernet_kernel<<<1, 512, smem1, stream>>>(
        p, W_ih1, W_hh1, b_ih1, b_hh1, W_ih2, W_hh2, b_ih2, b_hh2, fc1, wA);

    size_t smem2 = (size_t)IN_TILE_HALFS * 2 + (size_t)WA_HALFS * 2;   // 110080
    conv_wmma_kernel<<<dim3(224, 64), CONV_THREADS, smem2, stream>>>(input, wA, out);
}
